// Alignment_29532195127900
// MI455X (gfx1250) — matrix operand order repaired and checked by reading, not compiled, not run on hardware
//
#include <hip/hip_runtime.h>
#include <math.h>

// ---------------------------------------------------------------------------
// Decomposable-attention align step for MI455X (gfx1250, wave32, WMMA).
// B=64, L=D=A=512. All GEMMs run as bf16 WMMA (f32 accumulate).
// ---------------------------------------------------------------------------

typedef __attribute__((ext_vector_type(16))) __bf16 v16bf;
typedef __attribute__((ext_vector_type(8)))  __bf16 v8bf;
typedef __attribute__((ext_vector_type(4)))  __bf16 v4bf;
typedef __attribute__((ext_vector_type(8)))  float  v8f;

#define LQ    512                      // L == D == A
#define NB    64                       // batch
#define MATE  ((size_t)LQ * LQ)        // elements per 512x512 matrix
#define BLK   128                      // block tile M == N
#define BK    32                       // K step (bf16 WMMA K)
#define NK    (LQ / BK)                // 16 K-steps
#define LDST  40                       // padded LDS K-stride (bf16 elems, 80B rows)

// ---- branch-free fast tanh: (e^2x - 1) / (e^2x + 1) via v_exp_f32/v_rcp ----
static __device__ __forceinline__ float fast_tanh(float x) {
    x = fminf(9.0f, fmaxf(-9.0f, x));                       // keep exp2 finite
    const float t = __builtin_amdgcn_exp2f(x * 2.885390081777927f); // 2*log2(e)
    return (t - 1.0f) * __builtin_amdgcn_rcpf(t + 1.0f);
}

// ---- WMMA fragment loads (per CDNA5 ISA §7.12.2 layouts, wave32) -----------

// A fragment 16x32 bf16: lanes 0-15 row M=lane, K in v[0..3]={0..7}, v[4..7]={16..23};
// lanes 16-31 same rows, K {8..15} and {24..31}.
static __device__ __forceinline__ v16bf frag_a(const __bf16* As, int m0, int lane) {
    const int r = lane & 15, hf = lane >> 4;
    const __bf16* base = As + (size_t)(m0 + r) * LDST;
    union { v16bf v; v8bf h[2]; } u;
    u.h[0] = *(const v8bf*)(base + hf * 8);
    u.h[1] = *(const v8bf*)(base + 16 + hf * 8);
    return u.v;
}

// B fragment 32x16 bf16 from LDS stored [N][K]: lane (n = lane&15) reads its
// column's K {0..7},{16..23} (lanes 0-15) or {8..15},{24..31} (lanes 16-31).
static __device__ __forceinline__ v16bf frag_b(const __bf16* Bs, int n0, int lane) {
    const int c = lane & 15, hf = lane >> 4;
    const __bf16* base = Bs + (size_t)(n0 + c) * LDST;
    union { v16bf v; v8bf h[2]; } u;
    u.h[0] = *(const v8bf*)(base + hf * 8);
    u.h[1] = *(const v8bf*)(base + 16 + hf * 8);
    return u.v;
}

// ---------------------------------------------------------------------------
// Generic batched 512x512x512 GEMM, 128x128 block tiles, 8 waves per block,
// register-prefetch double-buffered LDS staging (one barrier per K-step).
//   A_BF16 : A operand stored bf16 (else f32 -> convert while staging)
//   A_T    : A element (m,k) at stored[k*512 + m]   (attn^T case)
//   B_BF16 : B operand stored bf16
//   B_NK   : B element (k,n) at stored[n*512 + k]   (direct row copy)
//   TANHO  : apply tanh, store bf16 (projection), else store f32
// ---------------------------------------------------------------------------
template<bool A_BF16, bool A_T, bool B_BF16, bool B_NK, bool TANHO>
__global__ __launch_bounds__(256) void gemm512(
    const void* __restrict__ Ap, size_t aBatch,
    const void* __restrict__ Bp, size_t bBatch,
    void* __restrict__ Op, size_t oBatch)
{
    __shared__ __bf16 As[2][BLK * LDST];
    __shared__ __bf16 Bs[2][BLK * LDST];

    const int b    = blockIdx.z;
    const int m0   = blockIdx.x * BLK;
    const int n0   = blockIdx.y * BLK;
    const int tid  = threadIdx.x;
    const int lane = tid & 31;
    const int wid  = tid >> 5;
    const int wm   = (wid & 3) * 32;   // 4 waves span 128 rows (2 subtiles each)
    const int wn   = (wid >> 2) * 64;  // 2 waves span 128 cols (4 subtiles each)

    const __bf16* Ab = A_BF16 ? (const __bf16*)Ap + aBatch * b : (const __bf16*)nullptr;
    const float*  Af = A_BF16 ? (const float*)nullptr : (const float*)Ap + aBatch * b;
    const __bf16* Bb = B_BF16 ? (const __bf16*)Bp + bBatch * b : (const __bf16*)nullptr;
    const float*  Bf = B_BF16 ? (const float*)nullptr : (const float*)Bp + bBatch * b;

    // ---- per-thread prefetch registers (16 elements of A, 16 of B) -------
    float4 paf[4]; uint4 pab[2];
    float4 pbf[4]; uint4 pbb[2];

    auto loadA = [&](int k0) {
        if constexpr (!A_T) {
            const int r = tid >> 1, c = (tid & 1) * 16;
            if constexpr (A_BF16) {
                const uint4* s = (const uint4*)(Ab + (size_t)(m0 + r) * LQ + k0 + c);
                pab[0] = s[0]; pab[1] = s[1];
            } else {
                const float4* s = (const float4*)(Af + (size_t)(m0 + r) * LQ + k0 + c);
                paf[0] = s[0]; paf[1] = s[1]; paf[2] = s[2]; paf[3] = s[3];
            }
        } else {
            const int kk = tid >> 3, c = (tid & 7) * 16;
            if constexpr (A_BF16) {
                const uint4* s = (const uint4*)(Ab + (size_t)(k0 + kk) * LQ + m0 + c);
                pab[0] = s[0]; pab[1] = s[1];
            } else {
                const float4* s = (const float4*)(Af + (size_t)(k0 + kk) * LQ + m0 + c);
                paf[0] = s[0]; paf[1] = s[1]; paf[2] = s[2]; paf[3] = s[3];
            }
        }
    };
    auto storeA = [&](__bf16* dst) {
        if constexpr (!A_T) {
            const int r = tid >> 1, c = (tid & 1) * 16;
            __bf16* d = dst + r * LDST + c;
            if constexpr (A_BF16) {
                *(uint4*)d = pab[0]; *(uint4*)(d + 8) = pab[1];
            } else {
                const float* f = (const float*)paf;
                v8bf o0, o1;
#pragma unroll
                for (int i = 0; i < 8; ++i) { o0[i] = (__bf16)f[i]; o1[i] = (__bf16)f[i + 8]; }
                *(v8bf*)d = o0; *(v8bf*)(d + 8) = o1;
            }
        } else {
            const int kk = tid >> 3, c = (tid & 7) * 16;
            if constexpr (A_BF16) {
                const __bf16* f = (const __bf16*)pab;
#pragma unroll
                for (int i = 0; i < 16; ++i) dst[(c + i) * LDST + kk] = f[i];
            } else {
                const float* f = (const float*)paf;
#pragma unroll
                for (int i = 0; i < 16; ++i) dst[(c + i) * LDST + kk] = (__bf16)f[i];
            }
        }
    };
    auto loadB = [&](int k0) {
        if constexpr (B_NK) {
            const int r = tid >> 1, c = (tid & 1) * 16;
            if constexpr (B_BF16) {
                const uint4* s = (const uint4*)(Bb + (size_t)(n0 + r) * LQ + k0 + c);
                pbb[0] = s[0]; pbb[1] = s[1];
            } else {
                const float4* s = (const float4*)(Bf + (size_t)(n0 + r) * LQ + k0 + c);
                pbf[0] = s[0]; pbf[1] = s[1]; pbf[2] = s[2]; pbf[3] = s[3];
            }
        } else {
            const int kk = tid >> 3, c = (tid & 7) * 16;
            if constexpr (B_BF16) {
                const uint4* s = (const uint4*)(Bb + (size_t)(k0 + kk) * LQ + n0 + c);
                pbb[0] = s[0]; pbb[1] = s[1];
            } else {
                const float4* s = (const float4*)(Bf + (size_t)(k0 + kk) * LQ + n0 + c);
                pbf[0] = s[0]; pbf[1] = s[1]; pbf[2] = s[2]; pbf[3] = s[3];
            }
        }
    };
    auto storeB = [&](__bf16* dst) {
        if constexpr (B_NK) {
            const int r = tid >> 1, c = (tid & 1) * 16;
            __bf16* d = dst + r * LDST + c;
            if constexpr (B_BF16) {
                *(uint4*)d = pbb[0]; *(uint4*)(d + 8) = pbb[1];
            } else {
                const float* f = (const float*)pbf;
                v8bf o0, o1;
#pragma unroll
                for (int i = 0; i < 8; ++i) { o0[i] = (__bf16)f[i]; o1[i] = (__bf16)f[i + 8]; }
                *(v8bf*)d = o0; *(v8bf*)(d + 8) = o1;
            }
        } else {
            const int kk = tid >> 3, c = (tid & 7) * 16;
            if constexpr (B_BF16) {
                const __bf16* f = (const __bf16*)pbb;
#pragma unroll
                for (int i = 0; i < 16; ++i) dst[(c + i) * LDST + kk] = f[i];
            } else {
                const float* f = (const float*)pbf;
#pragma unroll
                for (int i = 0; i < 16; ++i) dst[(c + i) * LDST + kk] = (__bf16)f[i];
            }
        }
    };

    // prologue: stage tile 0
    loadA(0); loadB(0);
    storeA(As[0]); storeB(Bs[0]);

    v8f acc[2][4] = {};

    for (int k = 0; k < NK; ++k) {
        __syncthreads();                       // buffer k&1 fully staged
        if (k + 1 < NK) { loadA((k + 1) * BK); loadB((k + 1) * BK); } // prefetch
        const __bf16* Ac = As[k & 1];
        const __bf16* Bc = Bs[k & 1];
        v16bf av[2], bv[4];
#pragma unroll
        for (int i = 0; i < 2; ++i) av[i] = frag_a(Ac, wm + i * 16, lane);
#pragma unroll
        for (int j = 0; j < 4; ++j) bv[j] = frag_b(Bc, wn + j * 16, lane);
#pragma unroll
        for (int i = 0; i < 2; ++i)
#pragma unroll
            for (int j = 0; j < 4; ++j)
                acc[i][j] = __builtin_amdgcn_wmma_f32_16x16x32_bf16(
                    false, av[i], false, bv[j], (short)0, acc[i][j], false, false);
        if (k + 1 < NK) { storeA(As[(k + 1) & 1]); storeB(Bs[(k + 1) & 1]); }
    }

    // ---- epilogue: C/D layout VGPR j -> row j (lanes 0-15) / 8+j (16-31) --
    const int col = lane & 15, hf = lane >> 4;
    if constexpr (TANHO) {
        __bf16* out = (__bf16*)Op + oBatch * b;
#pragma unroll
        for (int i = 0; i < 2; ++i)
#pragma unroll
            for (int j = 0; j < 4; ++j)
#pragma unroll
                for (int e = 0; e < 8; ++e)
                    out[(size_t)(m0 + wm + i * 16 + hf * 8 + e) * LQ +
                        (n0 + wn + j * 16 + col)] = (__bf16)fast_tanh(acc[i][j][e]);
    } else {
        float* out = (float*)Op + oBatch * b;
#pragma unroll
        for (int i = 0; i < 2; ++i)
#pragma unroll
            for (int j = 0; j < 4; ++j)
#pragma unroll
                for (int e = 0; e < 8; ++e)
                    out[(size_t)(m0 + wm + i * 16 + hf * 8 + e) * LQ +
                        (n0 + wn + j * 16 + col)] = acc[i][j][e];
    }
}

// ---------------------------------------------------------------------------
// Scores GEMM: e = F_p @ F_h^T. Both operands bf16 with direct (no-transpose)
// staging -> use gfx1250 async global->LDS copies (no VGPR round trip),
// pipelined one tile ahead on ASYNCcnt. IOFFSET shifts LDS & global together.
// ---------------------------------------------------------------------------
__global__ __launch_bounds__(256) void gemm512_scores_async(
    const __bf16* __restrict__ Fp, const __bf16* __restrict__ Fh,
    float* __restrict__ Eo)
{
    __shared__ __bf16 As[2][BLK * LDST];
    __shared__ __bf16 Bs[2][BLK * LDST];

    const int b    = blockIdx.z;
    const int m0   = blockIdx.x * BLK;
    const int n0   = blockIdx.y * BLK;
    const int tid  = threadIdx.x;
    const int lane = tid & 31;
    const int wid  = tid >> 5;
    const int wm   = (wid & 3) * 32;
    const int wn   = (wid >> 2) * 64;

    const __bf16* A  = Fp + MATE * b;
    const __bf16* Bm = Fh + MATE * b;
    const int r = tid >> 1, c = (tid & 1) * 16;   // each thread copies 32B of A + 32B of B

    auto stage = [&](int k0, int buf) {
        const __bf16* ga = A  + (size_t)(m0 + r) * LQ + k0 + c;
        const __bf16* gb = Bm + (size_t)(n0 + r) * LQ + k0 + c;
        unsigned la = (unsigned)(uintptr_t)&As[buf][r * LDST + c];
        unsigned lb = (unsigned)(uintptr_t)&Bs[buf][r * LDST + c];
        asm volatile(
            "global_load_async_to_lds_b128 %0, %1, off\n\t"
            "global_load_async_to_lds_b128 %0, %1, off offset:16\n\t"
            "global_load_async_to_lds_b128 %2, %3, off\n\t"
            "global_load_async_to_lds_b128 %2, %3, off offset:16"
            :: "v"(la), "v"((unsigned long long)(uintptr_t)ga),
               "v"(lb), "v"((unsigned long long)(uintptr_t)gb)
            : "memory");
    };

    stage(0, 0);
    v8f acc[2][4] = {};

    for (int k = 0; k < NK; ++k) {
        asm volatile("s_wait_asynccnt 0x0" ::: "memory");   // tile k landed in LDS
        __syncthreads();                                    // all waves' copies + prior reads done
        if (k + 1 < NK) stage((k + 1) * BK, (k + 1) & 1);   // overlap with WMMAs below
        const __bf16* Ac = As[k & 1];
        const __bf16* Bc = Bs[k & 1];
        v16bf av[2], bv[4];
#pragma unroll
        for (int i = 0; i < 2; ++i) av[i] = frag_a(Ac, wm + i * 16, lane);
#pragma unroll
        for (int j = 0; j < 4; ++j) bv[j] = frag_b(Bc, wn + j * 16, lane);
#pragma unroll
        for (int i = 0; i < 2; ++i)
#pragma unroll
            for (int j = 0; j < 4; ++j)
                acc[i][j] = __builtin_amdgcn_wmma_f32_16x16x32_bf16(
                    false, av[i], false, bv[j], (short)0, acc[i][j], false, false);
    }

    float* out = Eo + MATE * b;
    const int col = lane & 15, hf = lane >> 4;
#pragma unroll
    for (int i = 0; i < 2; ++i)
#pragma unroll
        for (int j = 0; j < 4; ++j)
#pragma unroll
            for (int e = 0; e < 8; ++e)
                out[(size_t)(m0 + wm + i * 16 + hf * 8 + e) * LQ +
                    (n0 + wn + j * 16 + col)] = acc[i][j][e];
}

// ---------------------------------------------------------------------------
// Row softmax: one wave32 per 512-wide row; 16 f32 per lane; shfl reductions.
// ---------------------------------------------------------------------------
__global__ __launch_bounds__(256) void softmax_rows(const float* __restrict__ e,
                                                    __bf16* __restrict__ attn)
{
    const int row  = blockIdx.x * 8 + (threadIdx.x >> 5);
    const int lane = threadIdx.x & 31;
    const float4* er = (const float4*)(e + (size_t)row * LQ);

    float4 v[4];
    float mx = -3.402823466e38f;
#pragma unroll
    for (int i = 0; i < 4; ++i) {
        v[i] = er[i * 32 + lane];
        mx = fmaxf(mx, fmaxf(fmaxf(v[i].x, v[i].y), fmaxf(v[i].z, v[i].w)));
    }
#pragma unroll
    for (int d = 16; d; d >>= 1) mx = fmaxf(mx, __shfl_xor(mx, d, 32));

    float s = 0.f;
#pragma unroll
    for (int i = 0; i < 4; ++i) {
        v[i].x = __builtin_amdgcn_exp2f((v[i].x - mx) * 1.4426950408889634f);
        v[i].y = __builtin_amdgcn_exp2f((v[i].y - mx) * 1.4426950408889634f);
        v[i].z = __builtin_amdgcn_exp2f((v[i].z - mx) * 1.4426950408889634f);
        v[i].w = __builtin_amdgcn_exp2f((v[i].w - mx) * 1.4426950408889634f);
        s += v[i].x + v[i].y + v[i].z + v[i].w;
    }
#pragma unroll
    for (int d = 16; d; d >>= 1) s += __shfl_xor(s, d, 32);
    const float inv = __builtin_amdgcn_rcpf(s);

    __bf16* ar = attn + (size_t)row * LQ;
#pragma unroll
    for (int i = 0; i < 4; ++i) {
        v4bf o;
        o[0] = (__bf16)(v[i].x * inv);  o[1] = (__bf16)(v[i].y * inv);
        o[2] = (__bf16)(v[i].z * inv);  o[3] = (__bf16)(v[i].w * inv);
        *(v4bf*)(ar + (size_t)(i * 32 + lane) * 4) = o;
    }
}

// ---------------------------------------------------------------------------
// Workspace layout (bytes):
//   [0,         67108864)  F  : bf16 [128][512][512] (F_p then F_h)
//   [67108864, 134217728)  e  : f32  [64][512][512]
//   attn bf16 [64][512][512] overwrites the F_p region after scores are done.
// ---------------------------------------------------------------------------
extern "C" void kernel_launch(void* const* d_in, const int* in_sizes, int n_in,
                              void* d_out, int out_size, void* d_ws, size_t ws_size,
                              hipStream_t stream)
{
    (void)in_sizes; (void)n_in; (void)out_size; (void)ws_size;

    const float* P = (const float*)d_in[0];   // premises   [64,512,512] f32
    const float* H = (const float*)d_in[1];   // hypotheses [64,512,512] f32
    const float* W = (const float*)d_in[2];   // W_F        [512,512]   f32
    float* out = (float*)d_out;               // betas then alphas, f32

    __bf16* F    = (__bf16*)d_ws;
    float*  Ebuf = (float*)((char*)d_ws + (size_t)2 * NB * MATE * sizeof(__bf16));
    __bf16* attn = (__bf16*)d_ws;             // reuse F_p region

    dim3 grid(LQ / BLK, LQ / BLK, NB), blk(256);

    // 1) F_p = tanh(P @ W), F_h = tanh(H @ W)
    gemm512<false, false, false, false, true><<<grid, blk, 0, stream>>>(
        P, MATE, W, 0, F, MATE);
    gemm512<false, false, false, false, true><<<grid, blk, 0, stream>>>(
        H, MATE, W, 0, F + (size_t)NB * MATE, MATE);

    // 2) e = F_p @ F_h^T  (async global->LDS staging)
    gemm512_scores_async<<<grid, blk, 0, stream>>>(
        F, F + (size_t)NB * MATE, Ebuf);

    // 3) attn = softmax(e) rows -> bf16 (overwrites dead F_p region)
    softmax_rows<<<dim3(NB * LQ / 8), blk, 0, stream>>>(Ebuf, attn);

    // 4) betas = attn @ H
    gemm512<true, false, false, false, false><<<grid, blk, 0, stream>>>(
        attn, MATE, H, MATE, out, MATE);

    // 5) alphas = attn^T @ P
    gemm512<true, true, false, false, false><<<grid, blk, 0, stream>>>(
        attn, MATE, P, MATE, out + (size_t)NB * MATE, MATE);
}
